// HeteroGNN_19241453486847
// MI455X (gfx1250) — matrix operand order repaired and checked by reading, not compiled, not run on hardware
//
#include <hip/hip_runtime.h>
#include <math.h>

// ---------------------------------------------------------------------------
// HGT forward, 2 layers, MI455X/gfx1250.
// rel.a / rel.m folded into K/V projection weights (one WMMA GEMM per type
// instead of per-edge einsum). Dense GEMMs: v_wmma_f32_16x16x32_bf16,
// 64x128 block tile, 32x32 per wave (4 accumulators), double-buffered LDS,
// wave32. Edge softmax-aggregation: encoded atomicMax + f32 atomicAdd.
// ---------------------------------------------------------------------------

#define EMB   768
#define HEADS 4
#define DH    192
#define NUSR  40000
#define NITM  20000
#define NEDGE 100000
#define NLAY  2

typedef __attribute__((ext_vector_type(16))) __bf16       v16bf;
typedef __attribute__((ext_vector_type(8)))  float        v8f;
typedef __attribute__((ext_vector_type(8)))  unsigned int v8u;

__device__ __forceinline__ unsigned short f2bf(float f) {
  unsigned int u = __builtin_bit_cast(unsigned int, f);
  u += 0x7FFFu + ((u >> 16) & 1u);           // round-to-nearest-even
  return (unsigned short)(u >> 16);
}
__device__ __forceinline__ unsigned int fenc(float x) {  // order-preserving
  unsigned int u = __builtin_bit_cast(unsigned int, x);
  return u ^ ((unsigned int)((int)u >> 31) | 0x80000000u);
}
__device__ __forceinline__ float fdec(unsigned int k) {
  unsigned int u = (k & 0x80000000u) ? (k ^ 0x80000000u) : ~k;
  return __builtin_bit_cast(float, u);
}

// ---------------------------------------------------------------------------
// bf16 WMMA GEMM: C[M,N](f32) = A[M,K](bf16) * B[K,N](bf16) (+ bias[N])
// grid = (ceil(M/64), N/BN); block = 256 = 8 waves (2x4), 32x32 per wave.
// N%BN==0, K%32==0. M guarded (clamped loads, predicated stores).
// ---------------------------------------------------------------------------
template <int BN>
__global__ __launch_bounds__(256) void gemm_bf16_wmma(
    const unsigned short* __restrict__ A, int lda,
    const unsigned short* __restrict__ B, int ldb,
    const float* __restrict__ bias,
    float* __restrict__ C, int ldc, int M, int K) {
  constexpr int BM = 64;
  constexpr int WN = BN / 4;   // wave N tile: 32 (BN=128) or 16 (BN=64)
  constexpr int NB = WN / 16;  // B fragments per wave
  __shared__ __align__(16) unsigned short As[2][BM * 32];
  __shared__ __align__(16) unsigned short Bs[2][32 * BN];

  const int m0   = blockIdx.x * BM;
  const int n0   = blockIdx.y * BN;
  const int tid  = threadIdx.x;
  const int lane = tid & 31;
  const int wave = tid >> 5;
  const int wm   = wave >> 2;  // 0..1 -> M offset wm*32
  const int wn   = wave & 3;   // 0..3 -> N offset wn*WN

  // global -> LDS staging (one b128 for A; BN/64 b128 for B per thread)
  const int arow = tid >> 2, acol = (tid & 3) << 3;       // 64 rows x 32
  const int brow = tid >> 3, bcol = (tid & 7) * (BN / 8); // 32 rows x BN
  const int asrow = (m0 + arow < M) ? (m0 + arow) : (M - 1);

  // fragment addressing (16-bit A 16x32 layout; B 32x16, C 16x16 f32)
  const int mr    = lane & 15;
  const int khalf = (lane >> 4) << 3;
  const int kb    = (lane & 15) + ((lane >> 4) << 4);

  v8f acc[2][NB];
#pragma unroll
  for (int a = 0; a < 2; ++a)
#pragma unroll
    for (int b = 0; b < NB; ++b) { v8f z = {}; acc[a][b] = z; }

  const int nsteps = K / 32;
  auto stage = [&](int buf, int k0) {
    *(uint4*)(&As[buf][arow * 32 + acol]) =
        *(const uint4*)(A + (size_t)asrow * lda + k0 + acol);
    *(uint4*)(&Bs[buf][brow * BN + bcol]) =
        *(const uint4*)(B + (size_t)(k0 + brow) * ldb + n0 + bcol);
    if constexpr (BN == 128) {
      *(uint4*)(&Bs[buf][brow * BN + bcol + 8]) =
          *(const uint4*)(B + (size_t)(k0 + brow) * ldb + n0 + bcol + 8);
    }
  };

  stage(0, 0);
  __syncthreads();

  for (int s = 0; s < nsteps; ++s) {
    const int cur = s & 1;
    if (s + 1 < nsteps) {
      stage(cur ^ 1, (s + 1) * 32);
      __builtin_prefetch(A + (size_t)asrow * lda + (s + 1) * 32 + acol, 0, 1);
      __builtin_prefetch(B + (size_t)((s + 1) * 32 + brow) * ldb + n0 + bcol, 0, 1);
    }

    const unsigned int* Asu = (const unsigned int*)As[cur];
    const unsigned int* Bsu = (const unsigned int*)Bs[cur];

    v16bf af[2];
#pragma unroll
    for (int fi = 0; fi < 2; ++fi) {
      const int rb = (wm * 32 + fi * 16 + mr) * 16;  // uint index, row stride 16
      v8u au;
#pragma unroll
      for (int i = 0; i < 4; ++i) au[i] = Asu[rb + ((khalf + 2 * i) >> 1)];
#pragma unroll
      for (int i = 0; i < 4; ++i) au[4 + i] = Asu[rb + ((16 + khalf + 2 * i) >> 1)];
      af[fi] = __builtin_bit_cast(v16bf, au);
    }
    v16bf bfr[NB];
#pragma unroll
    for (int fj = 0; fj < NB; ++fj) {
      const int bn = wn * WN + fj * 16 + mr;
      v8u bu;
#pragma unroll
      for (int j = 0; j < 8; ++j) {
        const int k = ((j & 4) << 2) + khalf + 2 * (j & 3);
        bu[j] = (unsigned int)Bs[cur][k * BN + bn] | ((unsigned int)Bs[cur][(k + 1) * BN + bn] << 16);
      }
      bfr[fj] = __builtin_bit_cast(v16bf, bu);
    }
#pragma unroll
    for (int fi = 0; fi < 2; ++fi)
#pragma unroll
      for (int fj = 0; fj < NB; ++fj)
        acc[fi][fj] = __builtin_amdgcn_wmma_f32_16x16x32_bf16(
            false, af[fi], false, bfr[fj], (short)0, acc[fi][fj], false, false);

    __syncthreads();
  }

  // C/D layout: VGPR i, lanes 0-15 -> M=i, N=lane; lanes 16-31 -> M=8+i.
#pragma unroll
  for (int fj = 0; fj < NB; ++fj) {
    const int col  = n0 + wn * WN + fj * 16 + (lane & 15);
    const float bv = bias ? bias[col] : 0.0f;
#pragma unroll
    for (int fi = 0; fi < 2; ++fi) {
      const int rb = m0 + wm * 32 + fi * 16 + ((lane >> 4) << 3);
#pragma unroll
      for (int i = 0; i < 8; ++i) {
        const int row = rb + i;
        if (row < M) C[(size_t)row * ldc + col] = acc[fi][fj][i] + bv;
      }
    }
  }
}

// --------------------------- elementwise kernels ---------------------------
__global__ void cvt_bf16_k(const float* __restrict__ x,
                           unsigned short* __restrict__ y, size_t n) {
  size_t i = (size_t)blockIdx.x * 256 + threadIdx.x;
  if (i < n) y[i] = f2bf(x[i]);
}

__global__ void fill_u32_k(unsigned int* __restrict__ p, unsigned int v, size_t n) {
  size_t i = (size_t)blockIdx.x * 256 + threadIdx.x;
  if (i < n) p[i] = v;
}

// b_out[h*192+f] = sum_d b_in[h*192+d] * rel[h,d,f]   (grid: 3x256 = 768)
__global__ void compose_bias_k(const float* __restrict__ b_in,
                               const float* __restrict__ rel,
                               float* __restrict__ b_out) {
  int t = blockIdx.x * 256 + threadIdx.x;
  if (t >= EMB) return;
  int h = t / DH, f = t % DH;
  const float* a  = rel + (size_t)h * DH * DH;
  const float* bi = b_in + h * DH;
  float s = 0.f;
  for (int d = 0; d < DH; ++d) s += bi[d] * a[d * DH + f];
  b_out[t] = s;
}

// ------------------------------ edge kernels -------------------------------
__global__ void edge_score_k(const float* __restrict__ q,
                             const float* __restrict__ kt,
                             const int* __restrict__ src, const int* __restrict__ dst,
                             const float* __restrict__ prior,
                             float* __restrict__ sc, int E) {
  int t = blockIdx.x * blockDim.x + threadIdx.x;
  if (t >= E * HEADS) return;
  int e = t >> 2, h = t & 3;
  const float4* qr = (const float4*)(q  + (size_t)dst[e] * EMB + h * DH);
  const float4* kr = (const float4*)(kt + (size_t)src[e] * EMB + h * DH);
  float a = 0.f;
#pragma unroll 4
  for (int i = 0; i < DH / 4; ++i) {
    float4 qa = qr[i], ka = kr[i];
    a += qa.x * ka.x + qa.y * ka.y + qa.z * ka.z + qa.w * ka.w;
  }
  sc[t] = a * prior[h] * 0.07216878364870323f;  // 1/sqrt(192)
}

__global__ void edge_max_k(const float* __restrict__ sc, const int* __restrict__ dst,
                           unsigned int* __restrict__ mx, int E) {
  int t = blockIdx.x * blockDim.x + threadIdx.x;
  if (t >= E * HEADS) return;
  int e = t >> 2, h = t & 3;
  atomicMax(&mx[(size_t)dst[e] * HEADS + h], fenc(sc[t]));
}

__global__ void edge_accum_k(const float* __restrict__ sc,
                             const unsigned int* __restrict__ mx,
                             const float* __restrict__ vt,
                             const int* __restrict__ src, const int* __restrict__ dst,
                             float* __restrict__ den, float* __restrict__ agg, int E) {
  int t = blockIdx.x * blockDim.x + threadIdx.x;
  if (t >= E * HEADS) return;
  int e = t >> 2, h = t & 3;
  int d = dst[e], s = src[e];
  float ex = expf(sc[t] - fdec(mx[(size_t)d * HEADS + h]));
  atomicAdd(&den[(size_t)d * HEADS + h], ex);
  const float4* vr = (const float4*)(vt + (size_t)s * EMB + h * DH);
  float* ar = agg + (size_t)d * EMB + h * DH;
#pragma unroll 4
  for (int i = 0; i < DH / 4; ++i) {
    float4 v = vr[i];
    atomicAdd(&ar[4 * i + 0], ex * v.x);
    atomicAdd(&ar[4 * i + 1], ex * v.y);
    atomicAdd(&ar[4 * i + 2], ex * v.z);
    atomicAdd(&ar[4 * i + 3], ex * v.w);
  }
}

__global__ void gelu_norm_k(const float* __restrict__ agg,
                            const float* __restrict__ den,
                            unsigned short* __restrict__ g, size_t ntot) {
  size_t i = (size_t)blockIdx.x * 256 + threadIdx.x;
  if (i >= ntot) return;
  int n = (int)(i / EMB), c = (int)(i % EMB), h = c / DH;
  float d = fmaxf(den[(size_t)n * HEADS + h], 1e-12f);
  float v = agg[i] / d;
  float ge = 0.5f * v * (1.0f + erff(v * 0.70710678118654752f));
  g[i] = f2bf(ge);
}

// out = LayerNorm(sigmoid(beta)*o + (1-sigmoid(beta))*h_prev)  (block per row)
__global__ __launch_bounds__(256) void skip_ln_k(
    const float* __restrict__ o, const float* __restrict__ hp,
    const float* __restrict__ betap, const float* __restrict__ g,
    const float* __restrict__ bb, float* __restrict__ out) {
  __shared__ float red[256];
  const int row = blockIdx.x, t = threadIdx.x;
  const size_t base = (size_t)row * EMB;
  const float beta = 1.0f / (1.0f + expf(-betap[0]));
  float xs[3], lsum = 0.f;
#pragma unroll
  for (int k = 0; k < 3; ++k) {
    int c = t + k * 256;
    float x = beta * o[base + c] + (1.0f - beta) * hp[base + c];
    xs[k] = x; lsum += x;
  }
  red[t] = lsum; __syncthreads();
  for (int s = 128; s > 0; s >>= 1) { if (t < s) red[t] += red[t + s]; __syncthreads(); }
  const float mu = red[0] * (1.0f / EMB);
  __syncthreads();
  float lv = 0.f;
#pragma unroll
  for (int k = 0; k < 3; ++k) { float d = xs[k] - mu; lv += d * d; }
  red[t] = lv; __syncthreads();
  for (int s = 128; s > 0; s >>= 1) { if (t < s) red[t] += red[t + s]; __syncthreads(); }
  const float r = rsqrtf(red[0] * (1.0f / EMB) + 1e-5f);
#pragma unroll
  for (int k = 0; k < 3; ++k) {
    int c = t + k * 256;
    out[base + c] = (xs[k] - mu) * r * g[c] + bb[c];
  }
}

// ---------------------------------------------------------------------------
extern "C" void kernel_launch(void* const* d_in, const int* in_sizes, int n_in,
                              void* d_out, int out_size, void* d_ws, size_t ws_size,
                              hipStream_t stream) {
  (void)in_sizes; (void)n_in; (void)out_size; (void)ws_size;
  const float* x_user = (const float*)d_in[0];
  const float* x_item = (const float*)d_in[1];
  const int* ui_src = (const int*)d_in[2];
  const int* ui_dst = (const int*)d_in[3];
  const int* iu_src = (const int*)d_in[4];
  const int* iu_dst = (const int*)d_in[5];
  float* outp = (float*)d_out;

  char* cur = (char*)d_ws;
  auto al = [&](size_t bytes) -> char* {
    char* p = cur; cur += (bytes + 255) & ~(size_t)255; return p;
  };

  unsigned short* actU = (unsigned short*)al((size_t)NUSR * EMB * 2);
  unsigned short* actI = (unsigned short*)al((size_t)NITM * EMB * 2);
  float* qU  = (float*)al((size_t)NUSR * EMB * 4);
  float* ktU = (float*)al((size_t)NUSR * EMB * 4);
  float* vtU = (float*)al((size_t)NUSR * EMB * 4);
  float* qI  = (float*)al((size_t)NITM * EMB * 4);
  float* ktI = (float*)al((size_t)NITM * EMB * 4);
  float* vtI = (float*)al((size_t)NITM * EMB * 4);
  float* aggU = (float*)al((size_t)NUSR * EMB * 4);
  float* aggI = (float*)al((size_t)NITM * EMB * 4);
  unsigned short* gU = (unsigned short*)al((size_t)NUSR * EMB * 2);
  unsigned short* gI = (unsigned short*)al((size_t)NITM * EMB * 2);
  float* oU = (float*)al((size_t)NUSR * EMB * 4);
  float* oI = (float*)al((size_t)NITM * EMB * 4);
  float* hU = (float*)al((size_t)NUSR * EMB * 4);
  float* hI = (float*)al((size_t)NITM * EMB * 4);
  unsigned int* mxU = (unsigned int*)al((size_t)NUSR * HEADS * 4);
  unsigned int* mxI = (unsigned int*)al((size_t)NITM * HEADS * 4);
  float* denU = (float*)al((size_t)NUSR * HEADS * 4);
  float* denI = (float*)al((size_t)NITM * HEADS * 4);
  float* scui = (float*)al((size_t)NEDGE * HEADS * 4);
  float* sciu = (float*)al((size_t)NEDGE * HEADS * 4);
  unsigned short* qwU = (unsigned short*)al((size_t)EMB * EMB * 2);
  unsigned short* qwI = (unsigned short*)al((size_t)EMB * EMB * 2);
  unsigned short* awU = (unsigned short*)al((size_t)EMB * EMB * 2);
  unsigned short* awI = (unsigned short*)al((size_t)EMB * EMB * 2);
  unsigned short* kwU = (unsigned short*)al((size_t)EMB * EMB * 2);
  unsigned short* kwI = (unsigned short*)al((size_t)EMB * EMB * 2);
  unsigned short* vwU = (unsigned short*)al((size_t)EMB * EMB * 2);
  unsigned short* vwI = (unsigned short*)al((size_t)EMB * EMB * 2);
  unsigned short* rAui = (unsigned short*)al((size_t)HEADS * DH * DH * 2);
  unsigned short* rMui = (unsigned short*)al((size_t)HEADS * DH * DH * 2);
  unsigned short* rAiu = (unsigned short*)al((size_t)HEADS * DH * DH * 2);
  unsigned short* rMiu = (unsigned short*)al((size_t)HEADS * DH * DH * 2);
  unsigned short* WkU = (unsigned short*)al((size_t)EMB * EMB * 2);
  unsigned short* WkI = (unsigned short*)al((size_t)EMB * EMB * 2);
  unsigned short* WvU = (unsigned short*)al((size_t)EMB * EMB * 2);
  unsigned short* WvI = (unsigned short*)al((size_t)EMB * EMB * 2);
  float* bkU = (float*)al(EMB * 4);
  float* bkI = (float*)al(EMB * 4);
  float* bvU = (float*)al(EMB * 4);
  float* bvI = (float*)al(EMB * 4);
  float* ctmp = (float*)al((size_t)EMB * EMB * 4);

  auto CVT = [&](const float* x, unsigned short* y, size_t n) {
    cvt_bf16_k<<<(unsigned)((n + 255) / 256), 256, 0, stream>>>(x, y, n);
  };
  auto FILL = [&](void* p, unsigned int v, size_t n) {
    fill_u32_k<<<(unsigned)((n + 255) / 256), 256, 0, stream>>>((unsigned int*)p, v, n);
  };
  auto GEMM = [&](const unsigned short* A, int lda, const unsigned short* B, int ldb,
                  const float* bias, float* C, int ldc, int M, int N, int K) {
    if (N % 128 == 0) {
      dim3 grid((M + 63) / 64, N / 128);
      gemm_bf16_wmma<128><<<grid, 256, 0, stream>>>(A, lda, B, ldb, bias, C, ldc, M, K);
    } else {
      dim3 grid((M + 63) / 64, N / 64);
      gemm_bf16_wmma<64><<<grid, 256, 0, stream>>>(A, lda, B, ldb, bias, C, ldc, M, K);
    }
  };
  auto COMPOSE = [&](const unsigned short* wbf, const unsigned short* rbf,
                     unsigned short* wout) {
    for (int h = 0; h < HEADS; ++h)
      GEMM(wbf + h * DH, EMB, rbf + (size_t)h * DH * DH, DH, nullptr,
           ctmp + h * DH, EMB, EMB, DH, DH);
    CVT(ctmp, wout, (size_t)EMB * EMB);
  };
  auto EDGE = [&](const float* qdst, const float* ktsrc, const float* vtsrc,
                  const int* src, const int* dst, const float* prior, float* sc,
                  unsigned int* mx, float* den, float* agg, int ndst) {
    FILL(mx, 0u, (size_t)ndst * HEADS);
    FILL(den, 0u, (size_t)ndst * HEADS);
    FILL(agg, 0u, (size_t)ndst * EMB);
    int nt = NEDGE * HEADS, nb = (nt + 255) / 256;
    edge_score_k<<<nb, 256, 0, stream>>>(qdst, ktsrc, src, dst, prior, sc, NEDGE);
    edge_max_k<<<nb, 256, 0, stream>>>(sc, dst, mx, NEDGE);
    edge_accum_k<<<nb, 256, 0, stream>>>(sc, mx, vtsrc, src, dst, den, agg, NEDGE);
  };

  for (int L = 0; L < NLAY; ++L) {
    // params assume JAX pytree (sorted-key) flattening: 26 leaves per layer:
    // 0:a_b[item] 1:a_b[user] 2:a_w[item] 3:a_w[user] 4:k_b[item] 5:k_b[user]
    // 6:k_w[item] 7:k_w[user] 8:ln_b 9:ln_g 10:q_b[item] 11:q_b[user]
    // 12:q_w[item] 13:q_w[user] 14:rel.iu.a 15:rel.iu.m 16:rel.iu.p
    // 17:rel.ui.a 18:rel.ui.m 19:rel.ui.p 20:skip[item] 21:skip[user]
    // 22:v_b[item] 23:v_b[user] 24:v_w[item] 25:v_w[user]
    const float* pp[26];
    for (int i = 0; i < 26; ++i) pp[i] = (const float*)d_in[6 + L * 26 + i];

    const float* hu = (L == 0) ? x_user : hU;
    const float* hi = (L == 0) ? x_item : hI;
    float* hou = (L == NLAY - 1) ? outp : hU;
    float* hoi = (L == NLAY - 1) ? (outp + (size_t)NUSR * EMB) : hI;

    // weights -> bf16
    CVT(pp[13], qwU, (size_t)EMB * EMB); CVT(pp[12], qwI, (size_t)EMB * EMB);
    CVT(pp[3],  awU, (size_t)EMB * EMB); CVT(pp[2],  awI, (size_t)EMB * EMB);
    CVT(pp[7],  kwU, (size_t)EMB * EMB); CVT(pp[6],  kwI, (size_t)EMB * EMB);
    CVT(pp[25], vwU, (size_t)EMB * EMB); CVT(pp[24], vwI, (size_t)EMB * EMB);
    CVT(pp[17], rAui, (size_t)HEADS * DH * DH); CVT(pp[18], rMui, (size_t)HEADS * DH * DH);
    CVT(pp[14], rAiu, (size_t)HEADS * DH * DH); CVT(pp[15], rMiu, (size_t)HEADS * DH * DH);

    // fold relation transforms into K/V weights (user feeds "ui", item "iu")
    COMPOSE(kwU, rAui, WkU); COMPOSE(vwU, rMui, WvU);
    COMPOSE(kwI, rAiu, WkI); COMPOSE(vwI, rMiu, WvI);
    compose_bias_k<<<3, 256, 0, stream>>>(pp[5], pp[17], bkU);
    compose_bias_k<<<3, 256, 0, stream>>>(pp[23], pp[18], bvU);
    compose_bias_k<<<3, 256, 0, stream>>>(pp[4], pp[14], bkI);
    compose_bias_k<<<3, 256, 0, stream>>>(pp[22], pp[15], bvI);

    // activations -> bf16
    CVT(hu, actU, (size_t)NUSR * EMB);
    CVT(hi, actI, (size_t)NITM * EMB);

    // projections (Q plain; K~/V~ with folded relation transform)
    GEMM(actU, EMB, qwU, EMB, pp[11], qU, EMB, NUSR, EMB, EMB);
    GEMM(actU, EMB, WkU, EMB, bkU,  ktU, EMB, NUSR, EMB, EMB);
    GEMM(actU, EMB, WvU, EMB, bvU,  vtU, EMB, NUSR, EMB, EMB);
    GEMM(actI, EMB, qwI, EMB, pp[10], qI, EMB, NITM, EMB, EMB);
    GEMM(actI, EMB, WkI, EMB, bkI,  ktI, EMB, NITM, EMB, EMB);
    GEMM(actI, EMB, WvI, EMB, bvI,  vtI, EMB, NITM, EMB, EMB);

    // relation "ui": user -> item
    EDGE(qI, ktU, vtU, ui_src, ui_dst, pp[19], scui, mxI, denI, aggI, NITM);
    gelu_norm_k<<<(unsigned)(((size_t)NITM * EMB + 255) / 256), 256, 0, stream>>>(
        aggI, denI, gI, (size_t)NITM * EMB);
    GEMM(gI, EMB, awI, EMB, pp[0], oI, EMB, NITM, EMB, EMB);
    skip_ln_k<<<NITM, 256, 0, stream>>>(oI, hi, pp[20], pp[9], pp[8], hoi);

    // relation "iu": item -> user
    EDGE(qU, ktI, vtI, iu_src, iu_dst, pp[16], sciu, mxU, denU, aggU, NUSR);
    gelu_norm_k<<<(unsigned)(((size_t)NUSR * EMB + 255) / 256), 256, 0, stream>>>(
        aggU, denU, gU, (size_t)NUSR * EMB);
    GEMM(gU, EMB, awU, EMB, pp[1], oU, EMB, NUSR, EMB, EMB);
    skip_ln_k<<<NUSR, 256, 0, stream>>>(oU, hu, pp[21], pp[9], pp[8], hou);
  }
}
